// FastAttention__89429809037713
// MI455X (gfx1250) — matrix operand order repaired and checked by reading, not compiled, not run on hardware
//
#include <hip/hip_runtime.h>
#include <math.h>

// ---------------------------------------------------------------------------
// Performer (FAVOR+) chunked causal linear attention for MI455X (gfx1250).
// All contractions run on v_wmma_f32_16x16x32_f16 (wave32 WMMA, fp32 accum).
// Problem is HBM-bound (~9 GFLOP vs ~170MB traffic @ 23.3 TB/s), so the
// design maximizes parallel workgroups (parallel chunked-scan form) and keeps
// every GEMM on the matrix pipe. Workspace requirement: ~169 MB.
// ---------------------------------------------------------------------------

typedef __attribute__((ext_vector_type(16))) _Float16 v16h;
typedef __attribute__((ext_vector_type(8)))  float    v8f;

constexpr int BB  = 2, HH = 8, NN = 4096, DD = 64, MM = 256, EE = 64;
constexpr int CH  = 128;                 // chunk length
constexpr int NCH = NN / CH;             // 32 chunks
constexpr int BH  = BB * HH;             // 16
constexpr long long BHN = (long long)BH * NN;   // 65536 rows

constexpr float DATA_NORM = 0.35355339059327373f;  // 64^-0.25
constexpr float RATIO     = 0.0625f;               // 256^-0.5
constexpr float KEPS      = 1e-4f;
constexpr float AEPS      = 1e-6f;

// workspace layout (in floats)
constexpr size_t OFF_QF   = 0;                               // BHN*MM
constexpr size_t OFF_KF   = OFF_QF + (size_t)BHN * MM;       // BHN*MM (raw scores, then kf in-place)
constexpr size_t OFF_KSS  = OFF_KF + (size_t)BHN * MM;       // BHN
constexpr size_t OFF_KSUM = OFF_KSS + (size_t)BHN;           // BH*NCH*MM
constexpr size_t OFF_KV   = OFF_KSUM + (size_t)BH * NCH * MM;// BH*NCH*MM*EE
constexpr size_t OFF_GMAX = OFF_KV + (size_t)BH * NCH * MM * EE; // 1 (as unsigned)

// output layout (floats): out[BHN*EE] | Z[BH*MM] | S[BH*MM*EE]
constexpr size_t ZOFF = (size_t)BHN * EE;
constexpr size_t SOFF = ZOFF + (size_t)BH * MM;

// ----------------------------- WMMA helpers --------------------------------
// 16-bit A fragment (16x32, M x K): lane&15 = M row; lane>>4 selects K half.
// VGPR v: K = (v<4 ? 2v : 8+2v) + 8*half  (pairs K,K+1 packed).
__device__ __forceinline__ v16h load_a(const _Float16* __restrict__ p, int stride, int koff) {
  const int lane = threadIdx.x & 31;
  const int hf = lane >> 4, r = lane & 15;
  v16h out;
#pragma unroll
  for (int vv = 0; vv < 8; ++vv) {
    const int kk = koff + ((vv < 4) ? (vv * 2) : (8 + vv * 2)) + hf * 8;
    out[2 * vv]     = p[r * stride + kk];
    out[2 * vv + 1] = p[r * stride + kk + 1];
  }
  return out;
}
// 16-bit B fragment (32x16, K x N), stored in LDS as [N][K] (K innermost):
// lane&15 = N col; VGPR v: K = (v<4 ? 2v : 8+2v) + 8*half (same K order as A).
__device__ __forceinline__ v16h load_b(const _Float16* __restrict__ p, int stride, int koff) {
  const int lane = threadIdx.x & 31;
  const int hf = lane >> 4, c = lane & 15;
  v16h out;
#pragma unroll
  for (int vv = 0; vv < 8; ++vv) {
    const int kk = koff + ((vv < 4) ? (vv * 2) : (8 + vv * 2)) + hf * 8;
    out[2 * vv]     = p[c * stride + kk];
    out[2 * vv + 1] = p[c * stride + kk + 1];
  }
  return out;
}
__device__ __forceinline__ v8f wmma_f16(v16h a, v16h b, v8f c) {
  return __builtin_amdgcn_wmma_f32_16x16x32_f16(false, a, false, b, (short)0, c, false, false);
}

// ordered-uint encoding of float for atomic max
__device__ __forceinline__ unsigned f2ord(float f) {
  unsigned u = __float_as_uint(f);
  return (u & 0x80000000u) ? ~u : (u | 0x80000000u);
}
__device__ __forceinline__ float ord2f(unsigned u) {
  unsigned v = (u & 0x80000000u) ? (u & 0x7fffffffu) : ~u;
  return __uint_as_float(v);
}

// ----------------------------- kernels -------------------------------------
__global__ void init_kernel(unsigned* gmax) { *gmax = f2ord(-3.4e38f); }

// Feature-map GEMM: dd = (data*norm) @ projT.  isQuery: row-max + exp -> out.
// !isQuery: raw dd -> out, scaled sumsq -> ssOut, global max -> gmax.
constexpr int FST = DD + 8;   // padded K stride (72) to avoid bank conflicts
__global__ void feat_kernel(const float* __restrict__ data, const float* __restrict__ proj,
                            float* __restrict__ out, float* __restrict__ ssOut,
                            unsigned* __restrict__ gmax, int isQuery) {
  __shared__ _Float16 dsm[CH * FST];     // scaled data rows  [128][72]
  __shared__ _Float16 psm[MM * FST];     // projection rows   [256][72]
  __shared__ float ss[CH];
  __shared__ unsigned smax;
  const int tid = threadIdx.x;
  const long long r0 = (long long)blockIdx.x * CH;

  if (tid == 0) smax = 0u;
  if (tid < CH) ss[tid] = 0.f;
  __syncthreads();
  {  // data: 128x64, each thread 32 elems; accumulate scaled sumsq per row
    const int row = tid >> 1, c0 = (tid & 1) * 32;
    const float* src = data + (size_t)(r0 + row) * DD + c0;
    float s = 0.f;
#pragma unroll
    for (int i = 0; i < 32; ++i) {
      float x = src[i] * DATA_NORM;
      s += x * x;
      dsm[row * FST + c0 + i] = (_Float16)x;
    }
    atomicAdd(&ss[row], s);
  }
  {  // proj: 256x64, each thread one row
    const float* src = proj + (size_t)tid * DD;
#pragma unroll
    for (int i = 0; i < DD; ++i) psm[tid * FST + i] = (_Float16)src[i];
  }
  __syncthreads();

  const int wave = tid >> 5, lane = tid & 31, hf = lane >> 4, idx = lane & 15;
  const int i0 = wave * 16;
  const v16h a0 = load_a(dsm + i0 * FST, FST, 0);
  const v16h a1 = load_a(dsm + i0 * FST, FST, 32);

  float vmax[8];
#pragma unroll
  for (int vv = 0; vv < 8; ++vv) vmax[vv] = -3.4e38f;
  for (int ct = 0; ct < MM / 16; ++ct) {   // pass 1: maxima
    v8f acc = {};
    acc = wmma_f16(a0, load_b(psm + ct * 16 * FST, FST, 0), acc);
    acc = wmma_f16(a1, load_b(psm + ct * 16 * FST, FST, 32), acc);
#pragma unroll
    for (int vv = 0; vv < 8; ++vv) vmax[vv] = fmaxf(vmax[vv], acc[vv]);
  }
#pragma unroll
  for (int m = 1; m <= 8; m <<= 1)       // row max within each 16-lane half
#pragma unroll
    for (int vv = 0; vv < 8; ++vv) vmax[vv] = fmaxf(vmax[vv], __shfl_xor(vmax[vv], m, 32));

  if (!isQuery) {
    float wm = vmax[0];
#pragma unroll
    for (int vv = 1; vv < 8; ++vv) wm = fmaxf(wm, vmax[vv]);
    wm = fmaxf(wm, __shfl_xor(wm, 16, 32));
    if (lane == 0) atomicMax(&smax, f2ord(wm));
  }
  for (int ct = 0; ct < MM / 16; ++ct) {   // pass 2: recompute + emit
    v8f acc = {};
    acc = wmma_f16(a0, load_b(psm + ct * 16 * FST, FST, 0), acc);
    acc = wmma_f16(a1, load_b(psm + ct * 16 * FST, FST, 32), acc);
#pragma unroll
    for (int vv = 0; vv < 8; ++vv) {
      const int row = i0 + vv + 8 * hf;
      const int col = ct * 16 + idx;
      float val = acc[vv];
      if (isQuery) val = RATIO * (__expf(val - 0.5f * ss[row] - vmax[vv]) + KEPS);
      out[(size_t)(r0 + row) * MM + col] = val;
    }
  }
  if (!isQuery) {
    if (tid < CH) ssOut[r0 + tid] = ss[tid];
    __syncthreads();
    if (tid == 0) atomicMax(gmax, smax);
  }
}

// kf exp pass with global stabilizer (in place over raw scores)
__global__ void exp_k_kernel(float* __restrict__ kf, const float* __restrict__ kss,
                             const unsigned* __restrict__ gmax) {
  const float stab = ord2f(*gmax);
  const size_t base = (size_t)blockIdx.x * 4096;
#pragma unroll
  for (int i = 0; i < 16; ++i) {
    const size_t id = base + (size_t)i * 256 + threadIdx.x;
    const size_t row = id >> 8;  // /MM
    kf[id] = RATIO * (__expf(kf[id] - 0.5f * kss[row] - stab) + KEPS);
  }
}

// Per-chunk K^T V [256x64] and column sums of K [256]  (parallel over bh,chunk)
constexpr int JST = CH + 8;  // 136: padded j-stride
__global__ void chunk_kv_kernel(const float* __restrict__ kf, const float* __restrict__ vsrc,
                                float* __restrict__ ckv, float* __restrict__ cksum) {
  extern __shared__ char sm[];
  _Float16* kt = (_Float16*)sm;                  // [MM][JST]  (A operand: m x j)
  _Float16* vt = kt + MM * JST;                  // [EE][JST]  (B operand: e rows, K=j)
  const int bh = blockIdx.x / NCH, c = blockIdx.x % NCH;
  const size_t nbase = (size_t)bh * NN + (size_t)c * CH;
  const int tid = threadIdx.x;
  for (int id = tid; id < CH * MM; id += 256) {
    const int j = id >> 8, m = id & 255;
    kt[m * JST + j] = (_Float16)kf[(nbase + j) * MM + m];
  }
  for (int id = tid; id < CH * EE; id += 256) {
    const int j = id >> 6, e = id & 63;
    vt[e * JST + j] = (_Float16)vsrc[(nbase + j) * EE + e];
  }
  __syncthreads();
  {  // K column sums
    float s = 0.f;
    for (int j = 0; j < CH; ++j) s += (float)kt[tid * JST + j];
    cksum[((size_t)bh * NCH + c) * MM + tid] = s;
  }
  const int wave = tid >> 5, lane = tid & 31, hf = lane >> 4, idx = lane & 15;
  for (int mt = wave; mt < MM / 16; mt += 8) {
    v16h a[4];
#pragma unroll
    for (int kk = 0; kk < 4; ++kk) a[kk] = load_a(kt + mt * 16 * JST, JST, kk * 32);
#pragma unroll
    for (int et = 0; et < 4; ++et) {
      v8f acc = {};
#pragma unroll
      for (int kk = 0; kk < 4; ++kk)
        acc = wmma_f16(a[kk], load_b(vt + et * 16 * JST, JST, kk * 32), acc);
#pragma unroll
      for (int vv = 0; vv < 8; ++vv) {
        const int m = mt * 16 + vv + 8 * hf, e = et * 16 + idx;
        ckv[(((size_t)bh * NCH + c) * MM + m) * EE + e] = acc[vv];
      }
    }
  }
}

// Exclusive prefix scan over chunks (in place); final totals -> Z, S outputs.
__global__ void scan_kernel(float* __restrict__ ckv, float* __restrict__ cksum,
                            float* __restrict__ outZ, float* __restrict__ outS) {
  const int bh = blockIdx.x, tid = threadIdx.x;
  float4 run[16];
#pragma unroll
  for (int u = 0; u < 16; ++u) { run[u].x = run[u].y = run[u].z = run[u].w = 0.f; }
  const size_t base = (size_t)bh * NCH * MM * EE;
  for (int c = 0; c < NCH; ++c) {
    float4* p = (float4*)(ckv + base + (size_t)c * MM * EE) + (size_t)tid * 16;
#pragma unroll
    for (int u = 0; u < 16; ++u) {
      float4 cur = p[u];
      p[u] = run[u];
      run[u].x += cur.x; run[u].y += cur.y; run[u].z += cur.z; run[u].w += cur.w;
    }
  }
  float4* so = (float4*)(outS + (size_t)bh * MM * EE) + (size_t)tid * 16;
#pragma unroll
  for (int u = 0; u < 16; ++u) so[u] = run[u];
  float rz = 0.f;
  const size_t kb = (size_t)bh * NCH * MM;
  for (int c = 0; c < NCH; ++c) {
    float cur = cksum[kb + (size_t)c * MM + tid];
    cksum[kb + (size_t)c * MM + tid] = rz;
    rz += cur;
  }
  outZ[(size_t)bh * MM + tid] = rz;
}

// Per-chunk output: A = QK^T (masked), out = A_mask*V + Q*S_prev, scaled by 1/D.
constexpr int QSTR = MM + 8;   // 264
constexpr int AFST = CH + 8;   // 136
__global__ void attn_out_kernel(const float* __restrict__ qf, const float* __restrict__ kf,
                                const float* __restrict__ vsrc, const float* __restrict__ ckv,
                                const float* __restrict__ cksum, float* __restrict__ out) {
  extern __shared__ char sm[];
  _Float16* Qc = (_Float16*)sm;          // [128][QSTR] A operand (K=m)
  _Float16* Kc = Qc + CH * QSTR;         // [128][QSTR] B operand rows=j (K=m)
  _Float16* Af = Kc + CH * QSTR;         // [128][AFST] masked A, f16 (K=j)
  _Float16* Vt = Af + CH * AFST;         // [64][AFST]  B operand rows=e (K=j)
  _Float16* St = Vt + EE * AFST;         // [64][QSTR]  B operand rows=e (K=m)
  float* Zp   = (float*)(St + EE * QSTR);// [256]
  float* Arow = Zp + MM;                 // [128]
  float* Dinv = Arow + CH;               // [128]

  const int bh = blockIdx.x / NCH, c = blockIdx.x % NCH;
  const size_t nbase = (size_t)bh * NN + (size_t)c * CH;
  const size_t cbase = ((size_t)bh * NCH + c);
  const int tid = threadIdx.x;

  for (int id = tid; id < CH * MM; id += 256) {
    const int i = id >> 8, m = id & 255;
    Qc[i * QSTR + m] = (_Float16)qf[(nbase + i) * MM + m];
    Kc[i * QSTR + m] = (_Float16)kf[(nbase + i) * MM + m];
  }
  for (int id = tid; id < CH * EE; id += 256) {
    const int j = id >> 6, e = id & 63;
    Vt[e * AFST + j] = (_Float16)vsrc[(nbase + j) * EE + e];
  }
  for (int id = tid; id < MM * EE; id += 256) {
    const int m = id >> 6, e = id & 63;
    St[e * QSTR + m] = (_Float16)ckv[(cbase * MM + m) * EE + e];
  }
  if (tid < MM) Zp[tid] = cksum[cbase * MM + tid];
  __syncthreads();

  const int wave = tid >> 5, lane = tid & 31, hf = lane >> 4, idx = lane & 15;
  const int i0 = wave * 16;
  v16h aq[8];
#pragma unroll
  for (int kk = 0; kk < 8; ++kk) aq[kk] = load_a(Qc + i0 * QSTR, QSTR, kk * 32);

  // GEMM1: A = Q K^T, causal mask (j<=i), row sums, stash masked A as f16
  float rs[8] = {0.f, 0.f, 0.f, 0.f, 0.f, 0.f, 0.f, 0.f};
  for (int ct = 0; ct < CH / 16; ++ct) {
    v8f acc = {};
#pragma unroll
    for (int kk = 0; kk < 8; ++kk)
      acc = wmma_f16(aq[kk], load_b(Kc + ct * 16 * QSTR, QSTR, kk * 32), acc);
#pragma unroll
    for (int vv = 0; vv < 8; ++vv) {
      const int i = i0 + vv + 8 * hf;
      const int j = ct * 16 + idx;
      const float val = (j <= i) ? acc[vv] : 0.f;
      rs[vv] += val;
      Af[i * AFST + j] = (_Float16)val;
    }
  }
#pragma unroll
  for (int m = 1; m <= 8; m <<= 1)
#pragma unroll
    for (int vv = 0; vv < 8; ++vv) rs[vv] += __shfl_xor(rs[vv], m, 32);
  if (idx == 0)
#pragma unroll
    for (int vv = 0; vv < 8; ++vv) Arow[i0 + vv + 8 * hf] = rs[vv];
  __syncthreads();

  // Denominators: D_i = q_i.Zprev + rowsum(A)_i + eps * sum_m q_i
  if (tid < CH) {
    float dz = 0.f, sq = 0.f;
    for (int m = 0; m < MM; ++m) {
      const float qv = (float)Qc[tid * QSTR + m];
      dz += qv * Zp[m];
      sq += qv;
    }
    Dinv[tid] = 1.0f / (dz + Arow[tid] + AEPS * sq);
  }
  __syncthreads();

  // GEMM2: out = Q*Sprev + Amask*V   (4 e-tiles per wave, shared accumulators)
  v8f zero = {};
  v8f acc[4];
#pragma unroll
  for (int et = 0; et < 4; ++et) acc[et] = zero;
#pragma unroll
  for (int kk = 0; kk < 8; ++kk) {
    const v16h a = aq[kk];
#pragma unroll
    for (int et = 0; et < 4; ++et)
      acc[et] = wmma_f16(a, load_b(St + et * 16 * QSTR, QSTR, kk * 32), acc[et]);
  }
#pragma unroll
  for (int kk = 0; kk < 4; ++kk) {
    const v16h a = load_a(Af + i0 * AFST, AFST, kk * 32);
#pragma unroll
    for (int et = 0; et < 4; ++et)
      acc[et] = wmma_f16(a, load_b(Vt + et * 16 * AFST, AFST, kk * 32), acc[et]);
  }
#pragma unroll
  for (int et = 0; et < 4; ++et)
#pragma unroll
    for (int vv = 0; vv < 8; ++vv) {
      const int i = i0 + vv + 8 * hf;
      const int e = et * 16 + idx;
      out[(nbase + i) * EE + e] = acc[et][vv] * Dinv[i];
    }
}

// ----------------------------- launcher ------------------------------------
extern "C" void kernel_launch(void* const* d_in, const int* in_sizes, int n_in,
                              void* d_out, int out_size, void* d_ws, size_t ws_size,
                              hipStream_t stream) {
  const float* q    = (const float*)d_in[0];
  const float* k    = (const float*)d_in[1];
  const float* v    = (const float*)d_in[2];
  const float* proj = (const float*)d_in[3];
  float* out = (float*)d_out;
  float* ws  = (float*)d_ws;

  float*    qf    = ws + OFF_QF;
  float*    kf    = ws + OFF_KF;
  float*    kss   = ws + OFF_KSS;
  float*    cksum = ws + OFF_KSUM;
  float*    ckv   = ws + OFF_KV;
  unsigned* gmax  = (unsigned*)(ws + OFF_GMAX);

  init_kernel<<<1, 1, 0, stream>>>(gmax);
  feat_kernel<<<(int)(BHN / CH), 256, 0, stream>>>(q, proj, qf, nullptr, gmax, 1);
  feat_kernel<<<(int)(BHN / CH), 256, 0, stream>>>(k, proj, kf, kss, gmax, 0);
  exp_k_kernel<<<(int)((BHN * MM) / 4096), 256, 0, stream>>>(kf, kss, gmax);

  const size_t smem_kv = (size_t)(MM * JST + EE * JST) * sizeof(_Float16);  // ~85 KB
  chunk_kv_kernel<<<BH * NCH, 256, smem_kv, stream>>>(kf, v, ckv, cksum);

  scan_kernel<<<BH, 256, 0, stream>>>(ckv, cksum, out + ZOFF, out + SOFF);

  const size_t smem_att = (size_t)(2 * CH * QSTR + CH * AFST + EE * AFST + EE * QSTR) *
                              sizeof(_Float16) +
                          (size_t)(MM + CH + CH) * sizeof(float);           // ~218 KB
  attn_out_kernel<<<BH * NCH, 256, smem_att, stream>>>(qf, kf, v, ckv, cksum, out);
}
